// DySPN_37056977830364
// MI455X (gfx1250) — matrix operand order repaired and checked by reading, not compiled, not run on hardware
//
#include <hip/hip_runtime.h>
#include <hip/hip_bf16.h>
#include <math.h>

#define B_    2
#define C_    64
#define H_    256
#define W_    1216
#define HP_   258
#define WP_   1218
#define KTOT  576          // Cin * 9
#define MTOT  80           // padded output channels
#define MUSE  78           // real output channels
#define NPIX  (B_*H_*W_)   // 622,592

typedef __attribute__((ext_vector_type(16))) __bf16 v16bf;
typedef __attribute__((ext_vector_type(8)))  __bf16 v8bf;
typedef __attribute__((ext_vector_type(8)))  float  v8f;

// ---------------------------------------------------------------------------
// Kernel 1: feature f32 NCHW -> bf16 NHWC with 1-pixel zero halo.
// featb[((b*HP_ + y)*WP_ + x)*C_ + c], interior y in [1,256], x in [1,1216].
// ---------------------------------------------------------------------------
__global__ void prep_feature(const float* __restrict__ feat, __bf16* __restrict__ out) {
    size_t i = (size_t)blockIdx.x * blockDim.x + threadIdx.x;
    const size_t total = (size_t)B_ * HP_ * WP_ * C_;
    if (i >= total) return;
    int c = (int)(i % C_);
    size_t r = i / C_;
    int x = (int)(r % WP_); r /= WP_;
    int y = (int)(r % HP_);
    int b = (int)(r / HP_);
    float v = 0.f;
    if (x >= 1 && x <= W_ && y >= 1 && y <= H_) {
        v = feat[(((size_t)b * C_ + c) * H_ + (y - 1)) * W_ + (x - 1)];
    }
    out[i] = (__bf16)v;
}

// ---------------------------------------------------------------------------
// Kernel 2: pack all 8 conv weights into bf16 A-matrix Wmat[80][576],
// K = (ky*3+kx)*64 + cin  (matches NHWC patch gather order).
// Rows: 0..23 aff7, 24..39 aff5, 40..47 aff3, 48..53 att7, 54..59 att5,
// 60..65 att3, 66..71 att1, 72..77 mask, 78..79 zero pad.
// sb[0..79]=scale, sb[80..159]=bias.
// ---------------------------------------------------------------------------
struct LayerPtrs { const float* w[8]; const float* s[8]; const float* b[8]; };

__global__ void prep_weights(LayerPtrs lp, __bf16* __restrict__ wmat, float* __restrict__ sb) {
    int gid = blockIdx.x * blockDim.x + threadIdx.x;
    if (gid >= MTOT * KTOT) return;
    int m = gid / KTOT, k = gid % KTOT;
    const int moff[8] = {0, 24, 40, 48, 54, 60, 66, 72};
    const int mcnt[8] = {24, 16, 8, 6, 6, 6, 6, 6};
    float v = 0.f;
    if (m < MUSE) {
        int l = 0;
        while (l < 7 && m >= moff[l] + mcnt[l]) l++;
        int oc  = m - moff[l];
        int tap = k / C_, cin = k % C_;
        int ky  = tap / 3, kx = tap % 3;
        v = lp.w[l][(((size_t)oc * C_ + cin) * 3 + ky) * 3 + kx];
        if (k == 0) { sb[m] = lp.s[l][oc]; sb[MTOT + m] = lp.b[l][oc]; }
    } else if (k == 0) {
        sb[m] = 0.f; sb[MTOT + m] = 0.f;
    }
    wmat[gid] = (__bf16)v;
}

// ---------------------------------------------------------------------------
// Kernel 3: implicit-GEMM conv via bf16 WMMA.
// Workgroup = 64 threads = 2 waves; each wave owns 2 N-tiles (32 consecutive
// pixels of one row) x all 5 M-tiles (80 couts) -> 10 independent WMMAs per
// K-step, each LDS A-fragment reused by 2 WMMAs.
// Weights (80x576 bf16, 92 KB) staged once in LDS. 18 K-steps of 32.
// A fragment (16x32 bf16): lane<16 holds K {0..15}, lane>=16 holds
// K {16..31} (contiguous, same order as B) -> two 16 B LDS chunks.
// B fragment (32x16 bf16): N = lane&15, K = (lane<16?0:16)+0..15 -> one
// contiguous 32 B NHWC channel load.
// Epilogue: scale/bias (+sigmoid for m>=48), planar f32 store.
// ---------------------------------------------------------------------------
__global__ __launch_bounds__(64) void conv_wmma(const __bf16* __restrict__ feat,
                                                const __bf16* __restrict__ wmat,
                                                const float* __restrict__ sb,
                                                float* __restrict__ convout) {
    extern __shared__ __bf16 Wl[];   // 80*576 bf16 = 92160 B
    const int tid = threadIdx.x;

    // cooperative weight stage: 5760 x uint4
    {
        const uint4* src = (const uint4*)wmat;
        uint4* dst = (uint4*)Wl;
        for (int i = tid; i < (MTOT * KTOT * 2) / 16; i += 64) dst[i] = src[i];
    }
    __syncthreads();

    const int lane = tid & 31;
    const int wave = tid >> 5;
    const int hi   = lane >> 4;                       // K-group select
    const int b = blockIdx.z, y = blockIdx.y;
    const int xl = blockIdx.x * 64 + wave * 32 + (lane & 15);  // N-tile 0 pixel

    v8f acc[5][2] = {};

    for (int kc = 0; kc < 18; ++kc) {
        const int tap = kc >> 1;
        const int c0  = (kc & 1) * 32;
        const int ky  = tap / 3, kx = tap - ky * 3;
        // halo coords: center pixel is (y+1, x+1); tap offset ky,kx in 0..2
        const size_t rowb = ((size_t)b * HP_ + (y + ky)) * WP_;
        if (kc + 1 < 18) {
            const int tap2 = (kc + 1) >> 1;
            const int ky2 = tap2 / 3, kx2 = tap2 - ky2 * 3;
            __builtin_prefetch(feat + (((size_t)b * HP_ + (y + ky2)) * WP_ + (xl + kx2)) * C_, 0, 3);
        }
        v16bf bf[2];
        #pragma unroll
        for (int nt = 0; nt < 2; ++nt) {
            const size_t boff = (rowb + (xl + nt * 16 + kx)) * C_ + c0 + hi * 16;
            bf[nt] = *(const v16bf*)(feat + boff);
        }
        #pragma unroll
        for (int mt = 0; mt < 5; ++mt) {
            const int m    = mt * 16 + (lane & 15);
            const int aoff = m * KTOT + kc * 32 + hi * 16;
            const v8bf lo = *(const v8bf*)(Wl + aoff);        // K {0..7}/{16..23}
            const v8bf hh = *(const v8bf*)(Wl + aoff + 8);    // K {8..15}/{24..31}
            const v16bf af = __builtin_shufflevector(lo, hh,
                0, 1, 2, 3, 4, 5, 6, 7, 8, 9, 10, 11, 12, 13, 14, 15);
            #pragma unroll
            for (int nt = 0; nt < 2; ++nt) {
                acc[mt][nt] = __builtin_amdgcn_wmma_f32_16x16x32_bf16(
                    false, af, false, bf[nt], (short)0, acc[mt][nt], false, false);
            }
        }
    }

    // epilogue: C/D layout -> VGPR r: lanes 0-15 M=r, lanes 16-31 M=r+8; N=lane&15
    #pragma unroll
    for (int nt = 0; nt < 2; ++nt) {
        const size_t pix = ((size_t)b * H_ + y) * W_ + (xl + nt * 16);
        #pragma unroll
        for (int mt = 0; mt < 5; ++mt) {
            #pragma unroll
            for (int r = 0; r < 8; ++r) {
                const int m = mt * 16 + r + hi * 8;
                if (m < MUSE) {
                    float yv = acc[mt][nt][r] * sb[m] + sb[MTOT + m];
                    if (m >= 48) yv = 1.f / (1.f + __expf(-yv));   // att* and mask
                    convout[(size_t)m * NPIX + pix] = yv;
                }
            }
        }
    }
}

// ---------------------------------------------------------------------------
// Kernel 4: one DySPN propagation iteration, fully fused.
// Each block: 16x16 output tile. Phase 1 computes normalized guides G[49],
// P[t]=G[t]*dt and g0 for the 22x22 halo into LDS (49*484 + 484 floats).
// Phase 2 gathers the 49 shifted taps, adds g0*d0, applies the masked blend.
// ---------------------------------------------------------------------------
__global__ __launch_bounds__(256) void dyspn_iter(const float* __restrict__ conv,
                                                  const float* __restrict__ d0,
                                                  const float* __restrict__ d00,
                                                  const float* __restrict__ dtin,
                                                  float* __restrict__ dtout,
                                                  int it) {
    extern __shared__ float sm[];
    const int HT = 22, NH = 484;
    float* sP  = sm;            // [49][484]
    float* sG0 = sm + 49 * NH;  // [484]

    // tap t -> (conv channel, att selector). ch for aff7:0..23, aff5:24..39,
    // aff3:40..47; ka: 0=att7, 1=att5, 2=att3, 3=att1 (center, v==1).
    static constexpr int CH[49] = {
        0, 1, 2, 3, 4, 5, 6, 7,
        24, 25, 26, 27, 28, 8, 9,
        29, 40, 41, 42, 30, 10, 11,
        31, 43, 0, 44, 32, 12, 13,
        33, 45, 46, 47, 34, 14, 15,
        35, 36, 37, 38, 39, 16, 17,
        18, 19, 20, 21, 22, 23 };
    static constexpr int KA[49] = {
        0, 0, 0, 0, 0, 0, 0, 0,
        1, 1, 1, 1, 1, 0, 0,
        1, 2, 2, 2, 1, 0, 0,
        1, 2, 3, 2, 1, 0, 0,
        1, 2, 2, 2, 1, 0, 0,
        1, 1, 1, 1, 1, 0, 0,
        0, 0, 0, 0, 0, 0 };

    const int b  = blockIdx.z;
    const int y0 = blockIdx.y * 16, x0 = blockIdx.x * 16;
    const int tid = threadIdx.x;

    // Phase 1: halo guide/P computation
    for (int li = tid; li < NH; li += 256) {
        const int hy = li / HT, hx = li % HT;
        const int gy = y0 + hy - 3, gx = x0 + hx - 3;
        if (gy < 0 || gy >= H_ || gx < 0 || gx >= W_) {
            #pragma unroll
            for (int t = 0; t < 49; ++t) sP[t * NH + li] = 0.f;
            sG0[li] = 0.f;
            continue;
        }
        const size_t p = ((size_t)b * H_ + gy) * W_ + gx;
        const float a7 = conv[(size_t)(48 + it) * NPIX + p];
        const float a5 = conv[(size_t)(54 + it) * NPIX + p];
        const float a3 = conv[(size_t)(60 + it) * NPIX + p];
        const float a1 = conv[(size_t)(66 + it) * NPIX + p];

        float v[49];
        float s7 = 0.f, s5 = 0.f, s3 = 0.f;
        #pragma unroll
        for (int t = 0; t < 49; ++t) {
            if (KA[t] == 3) { v[t] = 1.f; }
            else {
                v[t] = conv[(size_t)CH[t] * NPIX + p];
                const float av = fabsf(v[t]);
                if (KA[t] == 0)      s7 += av;
                else if (KA[t] == 1) s5 += av;
                else                 s3 += av;
            }
        }
        // att = sigmoid > 0, so |att*aff| = att*|aff|
        const float inv = 1.f / (a7 * s7 + a5 * s5 + a3 * s3 + a1);
        const float dtv = dtin[p];
        float gsum = 0.f;
        #pragma unroll
        for (int t = 0; t < 49; ++t) {
            const float aw = (KA[t] == 0) ? a7 : (KA[t] == 1) ? a5
                           : (KA[t] == 2) ? a3 : a1;
            const float g = aw * v[t] * inv;
            gsum += g;
            sP[t * NH + li] = g * dtv;
        }
        sG0[li] = 1.f - gsum;
    }
    __syncthreads();

    // Phase 2: gather. acc(y,x) = sum_t P[t](y+3-i, x+3-j)
    const int ly = tid >> 4, lx = tid & 15;
    const int gy = y0 + ly, gx = x0 + lx;
    const size_t p = ((size_t)b * H_ + gy) * W_ + gx;
    float acc = 0.f;
    #pragma unroll
    for (int t = 0; t < 49; ++t) {
        const int i = t / 7, j = t % 7;
        acc += sP[t * NH + (ly + 6 - i) * HT + (lx + 6 - j)];
    }
    const float d0v = d0[p], d00v = d00[p];
    const float dtn = acc + sG0[(ly + 3) * HT + (lx + 3)] * d0v;
    float mk = conv[(size_t)(72 + it) * NPIX + p];
    mk = (d00v > 0.f) ? mk : 0.f;        // mask_all = sigmoid(conv)*valid
    dtout[p] = mk * d00v + (1.f - mk) * dtn;
}

// ---------------------------------------------------------------------------
extern "C" void kernel_launch(void* const* d_in, const int* in_sizes, int n_in,
                              void* d_out, int out_size, void* d_ws, size_t ws_size,
                              hipStream_t stream) {
    const float* feature = (const float*)d_in[0];
    const float* d0  = (const float*)d_in[1];
    const float* d00 = (const float*)d_in[2];
    LayerPtrs lp;
    for (int l = 0; l < 8; ++l) {
        lp.w[l] = (const float*)d_in[3 + l * 3];
        lp.s[l] = (const float*)d_in[4 + l * 3];
        lp.b[l] = (const float*)d_in[5 + l * 3];
    }

    char* ws = (char*)d_ws;
    size_t off = 0;
    auto alloc = [&](size_t bytes) -> void* {
        void* p = ws + off;
        off = (off + bytes + 255) & ~(size_t)255;
        return p;
    };
    __bf16* featb = (__bf16*)alloc((size_t)B_ * HP_ * WP_ * C_ * 2);
    __bf16* wmat  = (__bf16*)alloc((size_t)MTOT * KTOT * 2);
    float*  sb    = (float*)alloc((size_t)2 * MTOT * 4);
    float*  convp = (float*)alloc((size_t)MUSE * NPIX * 4);
    float*  dtA   = (float*)alloc((size_t)NPIX * 4);
    float*  dtB   = (float*)alloc((size_t)NPIX * 4);

    {
        const size_t total = (size_t)B_ * HP_ * WP_ * C_;
        prep_feature<<<(unsigned)((total + 255) / 256), 256, 0, stream>>>(feature, featb);
    }
    prep_weights<<<(MTOT * KTOT + 255) / 256, 256, 0, stream>>>(lp, wmat, sb);
    {
        dim3 g(W_ / 64, H_, B_);
        conv_wmma<<<g, 64, MTOT * KTOT * 2, stream>>>(featb, wmat, sb, convp);
    }
    const float* din = d0;
    float* bufs[2] = {dtA, dtB};
    const size_t shm = (size_t)(49 * 484 + 484) * sizeof(float);
    for (int it = 0; it < 6; ++it) {
        float* dout = (it == 5) ? (float*)d_out : bufs[it & 1];
        dim3 g(W_ / 16, H_ / 16, B_);
        dyspn_iter<<<g, 256, shm, stream>>>(convp, d0, d00, din, dout, it);
        din = dout;
    }
}
